// ProteinGN_87608742904340
// MI455X (gfx1250) — compile-verified, matrix operand order checked by reading, not run on hardware
//
#include <hip/hip_runtime.h>
#include <math.h>

typedef __attribute__((ext_vector_type(16))) _Float16 v16h;
typedef __attribute__((ext_vector_type(8)))  float    v8f;

#define G_GRAPHS 256

// ---------------------------------------------------------------- utilities
__global__ void k_zero(float* __restrict__ p, long n) {
  long i = (long)blockIdx.x * blockDim.x + threadIdx.x;
  if (i < n) p[i] = 0.0f;
}

// edge_graph[e] = node_graph[senders[e]]; receiver in-degree; per-graph edge/node counts.
// All invariant across hops -> computed once.
__global__ void k_precompute(const int* __restrict__ senders,
                             const int* __restrict__ receivers,
                             const int* __restrict__ node_graph,
                             int*   __restrict__ egraph,
                             float* __restrict__ deg,
                             float* __restrict__ ecnt,
                             float* __restrict__ ncnt,
                             int E, int N) {
  int t = blockIdx.x * blockDim.x + threadIdx.x;
  if (t < E) {
    int s  = senders[t];
    int eg = node_graph[s];
    egraph[t] = eg;
    atomicAdd(&deg[receivers[t]], 1.0f);
    atomicAdd(&ecnt[eg], 1.0f);
  }
  if (t < N) {
    atomicAdd(&ncnt[node_graph[t]], 1.0f);
  }
}

// ------------------------------------------------------------ edge encoder
// e = relu(relu(edges@We1+be1)@We2+be2)   [E,16]->[E,4]->[E,8]
__global__ __launch_bounds__(256) void k_edge_encoder(
    const float* __restrict__ edges,
    const float* __restrict__ We1, const float* __restrict__ be1,
    const float* __restrict__ We2, const float* __restrict__ be2,
    float* __restrict__ e, int E) {
  int t = blockIdx.x * blockDim.x + threadIdx.x;
  if (t >= E) return;
  const float* x = edges + (size_t)t * 16;
  float xr[16];
#pragma unroll
  for (int k = 0; k < 16; k++) xr[k] = x[k];
  float h1[4];
#pragma unroll
  for (int j = 0; j < 4; j++) {
    float a = be1[j];
#pragma unroll
    for (int k = 0; k < 16; k++) a += xr[k] * We1[k * 4 + j];
    h1[j] = fmaxf(a, 0.0f);
  }
#pragma unroll
  for (int j = 0; j < 8; j++) {
    float a = be2[j];
#pragma unroll
    for (int k = 0; k < 4; k++) a += h1[k] * We2[k * 8 + j];
    e[(size_t)t * 8 + j] = fmaxf(a, 0.0f);
  }
}

// ------------------------------------------------------------ node encoder (WMMA)
// n = relu(relu(nodes@Wn1+bn1)@Wn2+bn2)   [N,128]->[N,32]->[N,16]
// One wave per 16-node tile. GEMM1: K=128 (4 k-steps of 32) x 2 N-tiles.
// GEMM2: one 16x16x32 WMMA after a D-layout -> A-layout shuffle via LDS.
// B operands are pre-swizzled into per-lane register layout in LDS so each
// fetch is one contiguous 32-byte v16h read (ds_load_b128 x2), not 16x u16.
__global__ __launch_bounds__(256) void k_node_encoder(
    const float* __restrict__ nodes,
    const float* __restrict__ Wn1, const float* __restrict__ bn1,
    const float* __restrict__ Wn2, const float* __restrict__ bn2,
    float* __restrict__ nOut, int N)
{
  __shared__ __align__(32) _Float16 sB1[4 * 2 * 32 * 16]; // [kb][ntile][lane][h] 8KB
  __shared__ __align__(32) _Float16 sB2[32 * 16];         // [lane][h]            1KB
  __shared__ __align__(16) float    sN1[8 * 16 * 32];     // per-wave 16x32 tile 16KB

  int tid = threadIdx.x;
  // Stage GEMM1 B swizzled: value(lane,h) = Wn1[K][nt*16 + lane%16],
  // K = kb*32 + (h<8 ? h : h+8) + (lane/16)*8   (16-bit B-layout)
  for (int i = tid; i < 4096; i += 256) {
    int h    = i & 15;
    int lane = (i >> 4) & 31;
    int nt   = (i >> 9) & 1;
    int kb   = (i >> 10) & 3;
    int K = kb * 32 + ((h < 8) ? h : h + 8) + (lane >> 4) * 8;
    sB1[i] = (_Float16)Wn1[K * 32 + nt * 16 + (lane & 15)];
  }
  // Stage GEMM2 B swizzled (single K-block of 32)
  for (int i = tid; i < 512; i += 256) {
    int h    = i & 15;
    int lane = (i >> 4) & 31;
    int K = ((h < 8) ? h : h + 8) + (lane >> 4) * 8;
    sB2[i] = (_Float16)Wn2[K * 16 + (lane & 15)];
  }
  __syncthreads();

  int wave = tid >> 5;
  int lane = tid & 31;
  int hi   = lane >> 4;     // half-wave select
  int ln   = lane & 15;

  long tileBase = ((long)blockIdx.x * 8 + wave) * 16;
  if (tileBase >= N) return;            // wave-uniform exit, EXEC stays all-1 for WMMA

  long rowA = tileBase + ln;            // branchless clamp for ragged last tile
  if (rowA > (long)N - 1) rowA = (long)N - 1;
  const float* aRow = nodes + rowA * 128;

  v8f acc0 = {};
  v8f acc1 = {};
#pragma unroll
  for (int kb = 0; kb < 4; kb++) {
    // A: two contiguous 8-float runs per lane -> global_load_b128 + cvt_pk
    v16h a;
    {
      const float4* ap = (const float4*)(aRow + kb * 32 + hi * 8);
      float4 f0 = ap[0], f1 = ap[1];        // K = kb*32+hi*8 .. +7   (halves 0..7)
      const float4* aq = (const float4*)(aRow + kb * 32 + 16 + hi * 8);
      float4 f2 = aq[0], f3 = aq[1];        // K = kb*32+16+hi*8 .. +7 (halves 8..15)
      a[0]=(_Float16)f0.x;  a[1]=(_Float16)f0.y;  a[2]=(_Float16)f0.z;  a[3]=(_Float16)f0.w;
      a[4]=(_Float16)f1.x;  a[5]=(_Float16)f1.y;  a[6]=(_Float16)f1.z;  a[7]=(_Float16)f1.w;
      a[8]=(_Float16)f2.x;  a[9]=(_Float16)f2.y;  a[10]=(_Float16)f2.z; a[11]=(_Float16)f2.w;
      a[12]=(_Float16)f3.x; a[13]=(_Float16)f3.y; a[14]=(_Float16)f3.z; a[15]=(_Float16)f3.w;
    }
    v16h b0 = *(const v16h*)&sB1[((kb * 2 + 0) * 32 + lane) * 16];
    v16h b1 = *(const v16h*)&sB1[((kb * 2 + 1) * 32 + lane) * 16];
    acc0 = __builtin_amdgcn_wmma_f32_16x16x32_f16(false, a, false, b0, (short)0, acc0, false, false);
    acc1 = __builtin_amdgcn_wmma_f32_16x16x32_f16(false, a, false, b1, (short)0, acc1, false, false);
  }

  // bias + relu, write D-layout (row = v + 8*hi, col = ln) into row-major LDS
  float* myN1 = sN1 + wave * (16 * 32);
  float bias0 = bn1[ln], bias1 = bn1[16 + ln];
#pragma unroll
  for (int v = 0; v < 8; v++) {
    int m = v + 8 * hi;
    myN1[m * 32 + ln]      = fmaxf(acc0[v] + bias0, 0.0f);
    myN1[m * 32 + 16 + ln] = fmaxf(acc1[v] + bias1, 0.0f);
  }
  asm volatile("s_wait_dscnt 0" ::: "memory");  // wave-local LDS RAW fence

  // GEMM2: A = myN1 (16x32). Per lane: two contiguous 8-float runs -> ds_load_b128.
  v16h a2;
  {
    const float4* rp = (const float4*)(myN1 + ln * 32);
    float4 f0 = rp[hi * 2], f1 = rp[hi * 2 + 1];       // K = hi*8 .. +7
    float4 f2 = rp[4 + hi * 2], f3 = rp[5 + hi * 2];   // K = 16+hi*8 .. +7
    a2[0]=(_Float16)f0.x;  a2[1]=(_Float16)f0.y;  a2[2]=(_Float16)f0.z;  a2[3]=(_Float16)f0.w;
    a2[4]=(_Float16)f1.x;  a2[5]=(_Float16)f1.y;  a2[6]=(_Float16)f1.z;  a2[7]=(_Float16)f1.w;
    a2[8]=(_Float16)f2.x;  a2[9]=(_Float16)f2.y;  a2[10]=(_Float16)f2.z; a2[11]=(_Float16)f2.w;
    a2[12]=(_Float16)f3.x; a2[13]=(_Float16)f3.y; a2[14]=(_Float16)f3.z; a2[15]=(_Float16)f3.w;
  }
  v16h b2 = *(const v16h*)&sB2[lane * 16];
  v8f acc2 = {};
  acc2 = __builtin_amdgcn_wmma_f32_16x16x32_f16(false, a2, false, b2, (short)0, acc2, false, false);

  float bias2 = bn2[ln];
#pragma unroll
  for (int v = 0; v < 8; v++) {
    long m = tileBase + v + 8 * hi;
    if (m < N) nOut[m * 16 + ln] = fmaxf(acc2[v] + bias2, 0.0f);
  }
}

// sum encoder node features per graph (for g init)
__global__ void k_node_gsum(const float* __restrict__ n, const int* __restrict__ node_graph,
                            float* __restrict__ gn_sum, int N) {
  int t = blockIdx.x * blockDim.x + threadIdx.x;
  if (t >= N) return;
  int gi = node_graph[t];
  const float* row = n + (size_t)t * 16;
#pragma unroll
  for (int c = 0; c < 16; c++) atomicAdd(&gn_sum[gi * 16 + c], row[c]);
}

// g = seg_mean(n) @ Wg2 + bg2   (no relu)
__global__ void k_g_init(const float* __restrict__ gn_sum, const float* __restrict__ ncnt,
                         const float* __restrict__ Wg2, const float* __restrict__ bg2,
                         float* __restrict__ g) {
  int j = blockIdx.x * blockDim.x + threadIdx.x;
  if (j >= G_GRAPHS) return;
  float inv = 1.0f / fmaxf(ncnt[j], 1.0f);
  float nm[16];
#pragma unroll
  for (int c = 0; c < 16; c++) nm[c] = gn_sum[j * 16 + c] * inv;
#pragma unroll
  for (int c = 0; c < 4; c++) {
    float a = bg2[c];
#pragma unroll
    for (int k = 0; k < 16; k++) a += nm[k] * Wg2[k * 4 + c];
    g[j * 4 + c] = a;
  }
}

// ---------------------------------------------------------------- hop kernels
__global__ __launch_bounds__(256) void k_edge_update(
    const int* __restrict__ senders, const int* __restrict__ receivers,
    const int* __restrict__ egraph,
    const float* __restrict__ n, const float* __restrict__ g,
    const float* __restrict__ He_e, const float* __restrict__ He_s,
    const float* __restrict__ He_g, const float* __restrict__ He_b,
    float* __restrict__ e, float* __restrict__ inc_sum, float* __restrict__ ge_sum, int E)
{
  __shared__ float sWe[8 * 8], sWs[16 * 8], sWg[4 * 8], sB[8];
  int tid = threadIdx.x;
  if (tid < 64)                 sWe[tid]       = He_e[tid];
  if (tid >= 64 && tid < 192)   sWs[tid - 64]  = He_s[tid - 64];
  if (tid >= 192 && tid < 224)  sWg[tid - 192] = He_g[tid - 192];
  if (tid >= 224 && tid < 232)  sB[tid - 224]  = He_b[tid - 224];
  __syncthreads();

  int t = blockIdx.x * 256 + tid;
  if (t >= E) return;
  float ein[8];
#pragma unroll
  for (int k = 0; k < 8; k++) ein[k] = e[(size_t)t * 8 + k];
  int s = senders[t], r = receivers[t], eg = egraph[t];
  float ns[16];
#pragma unroll
  for (int k = 0; k < 16; k++) ns[k] = n[(size_t)s * 16 + k];
  float gg[4];
#pragma unroll
  for (int k = 0; k < 4; k++) gg[k] = g[eg * 4 + k];

  float out[8];
#pragma unroll
  for (int j = 0; j < 8; j++) {
    float a = sB[j];
#pragma unroll
    for (int k = 0; k < 8; k++)  a += ein[k] * sWe[k * 8 + j];
#pragma unroll
    for (int k = 0; k < 16; k++) a += ns[k] * sWs[k * 8 + j];
#pragma unroll
    for (int k = 0; k < 4; k++)  a += gg[k] * sWg[k * 8 + j];
    out[j] = fmaxf(a, 0.0f);
  }
#pragma unroll
  for (int j = 0; j < 8; j++) e[(size_t)t * 8 + j] = out[j];
#pragma unroll
  for (int j = 0; j < 8; j++) atomicAdd(&inc_sum[(size_t)r * 8 + j], out[j]);
#pragma unroll
  for (int j = 0; j < 8; j++) atomicAdd(&ge_sum[eg * 8 + j], out[j]);
}

__global__ __launch_bounds__(256) void k_node_update(
    const int* __restrict__ node_graph,
    const float* __restrict__ inc_sum, const float* __restrict__ deg,
    const float* __restrict__ g,
    const float* __restrict__ Hn_n, const float* __restrict__ Hn_in,
    const float* __restrict__ Hn_g, const float* __restrict__ Hn_b,
    float* __restrict__ n, float* __restrict__ gn_sum, int N)
{
  __shared__ float sWn[16 * 16], sWi[8 * 16], sWg[4 * 16], sB[16];
  int tid = threadIdx.x;
  sWn[tid] = Hn_n[tid];                    // 256 values, one per thread
  if (tid < 128) sWi[tid] = Hn_in[tid];
  if (tid < 64)  sWg[tid] = Hn_g[tid];
  if (tid < 16)  sB[tid]  = Hn_b[tid];
  __syncthreads();

  int t = blockIdx.x * 256 + tid;
  if (t >= N) return;
  float ni[16];
#pragma unroll
  for (int k = 0; k < 16; k++) ni[k] = n[(size_t)t * 16 + k];
  float inv = 1.0f / fmaxf(deg[t], 1.0f);
  float inc[8];
#pragma unroll
  for (int k = 0; k < 8; k++) inc[k] = inc_sum[(size_t)t * 8 + k] * inv;
  int ng = node_graph[t];
  float gg[4];
#pragma unroll
  for (int k = 0; k < 4; k++) gg[k] = g[ng * 4 + k];

  float out[16];
#pragma unroll
  for (int j = 0; j < 16; j++) {
    float a = sB[j];
#pragma unroll
    for (int k = 0; k < 16; k++) a += ni[k] * sWn[k * 16 + j];
#pragma unroll
    for (int k = 0; k < 8; k++)  a += inc[k] * sWi[k * 16 + j];
#pragma unroll
    for (int k = 0; k < 4; k++)  a += gg[k] * sWg[k * 16 + j];
    out[j] = fmaxf(a, 0.0f);
  }
#pragma unroll
  for (int j = 0; j < 16; j++) n[(size_t)t * 16 + j] = out[j];
#pragma unroll
  for (int j = 0; j < 16; j++) atomicAdd(&gn_sum[ng * 16 + j], out[j]);
}

__global__ void k_global_update(
    const float* __restrict__ ge_sum, const float* __restrict__ gn_sum,
    const float* __restrict__ ecnt,   const float* __restrict__ ncnt,
    const float* __restrict__ Hg_e, const float* __restrict__ Hg_n,
    const float* __restrict__ Hg_g, const float* __restrict__ Hg_b,
    float* __restrict__ g)
{
  int j = blockIdx.x * blockDim.x + threadIdx.x;
  if (j >= G_GRAPHS) return;
  float einv = 1.0f / fmaxf(ecnt[j], 1.0f);
  float ninv = 1.0f / fmaxf(ncnt[j], 1.0f);
  float em[8], nm[16], go[4], out[4];
#pragma unroll
  for (int k = 0; k < 8; k++)  em[k] = ge_sum[j * 8 + k] * einv;
#pragma unroll
  for (int k = 0; k < 16; k++) nm[k] = gn_sum[j * 16 + k] * ninv;
#pragma unroll
  for (int k = 0; k < 4; k++)  go[k] = g[j * 4 + k];
#pragma unroll
  for (int c = 0; c < 4; c++) {
    float a = Hg_b[c];
#pragma unroll
    for (int k = 0; k < 8; k++)  a += em[k] * Hg_e[k * 4 + c];
#pragma unroll
    for (int k = 0; k < 16; k++) a += nm[k] * Hg_n[k * 4 + c];
#pragma unroll
    for (int k = 0; k < 4; k++)  a += go[k] * Hg_g[k * 4 + c];
    out[c] = fmaxf(a, 0.0f);
  }
#pragma unroll
  for (int c = 0; c < 4; c++) g[j * 4 + c] = out[c];
}

// ---------------------------------------------------------------- readouts
__global__ void k_readout(const float* __restrict__ n, const float* __restrict__ g,
                          const float* __restrict__ Rn, const float* __restrict__ Rn_b,
                          const float* __restrict__ Rg, const float* __restrict__ Rg_b,
                          float* __restrict__ out, int N)
{
  int t = blockIdx.x * blockDim.x + threadIdx.x;
  if (t < N) {
    float a = Rn_b[0];
#pragma unroll
    for (int k = 0; k < 16; k++) a += n[(size_t)t * 16 + k] * Rn[k];
    out[t] = 1.0f / (1.0f + expf(-a));
  } else if (t < N + G_GRAPHS) {
    int j = t - N;
    float a = Rg_b[0];
#pragma unroll
    for (int k = 0; k < 4; k++) a += g[j * 4 + k] * Rg[k];
    out[t] = 1.0f / (1.0f + expf(-a));
  }
}

// ---------------------------------------------------------------- launch
extern "C" void kernel_launch(void* const* d_in, const int* in_sizes, int n_in,
                              void* d_out, int out_size, void* d_ws, size_t ws_size,
                              hipStream_t stream) {
  const float* nodes      = (const float*)d_in[0];
  const float* edges      = (const float*)d_in[1];
  const int*   senders    = (const int*)d_in[2];
  const int*   receivers  = (const int*)d_in[3];
  const int*   node_graph = (const int*)d_in[4];
  const float* We1 = (const float*)d_in[5];
  const float* be1 = (const float*)d_in[6];
  const float* Wn1 = (const float*)d_in[7];
  const float* bn1 = (const float*)d_in[8];
  const float* We2 = (const float*)d_in[9];
  const float* be2 = (const float*)d_in[10];
  const float* Wn2 = (const float*)d_in[11];
  const float* bn2 = (const float*)d_in[12];
  const float* Wg2 = (const float*)d_in[13];
  const float* bg2 = (const float*)d_in[14];
  const float* He_e = (const float*)d_in[15];
  const float* He_s = (const float*)d_in[16];
  const float* He_g = (const float*)d_in[17];
  const float* He_b = (const float*)d_in[18];
  const float* Hn_n = (const float*)d_in[19];
  const float* Hn_in = (const float*)d_in[20];
  const float* Hn_g = (const float*)d_in[21];
  const float* Hn_b = (const float*)d_in[22];
  const float* Hg_e = (const float*)d_in[23];
  const float* Hg_n = (const float*)d_in[24];
  const float* Hg_g = (const float*)d_in[25];
  const float* Hg_b = (const float*)d_in[26];
  const float* Rn   = (const float*)d_in[27];
  const float* Rn_b = (const float*)d_in[28];
  const float* Rg   = (const float*)d_in[29];
  const float* Rg_b = (const float*)d_in[30];

  const int E = in_sizes[2];
  const int N = in_sizes[4];
  const int G = G_GRAPHS;

  // workspace layout (all invariant slices contiguous where we need bulk zeroing)
  float* ws     = (float*)d_ws;
  float* e      = ws;                           // [E,8]
  float* n      = e + (size_t)E * 8;            // [N,16]
  float* inc    = n + (size_t)N * 16;           // [N,8]
  float* deg    = inc + (size_t)N * 8;          // [N]
  int*   egraph = (int*)(deg + N);              // [E]
  float* g      = (float*)(egraph + E);         // [G,4]
  float* ge_sum = g + (size_t)G * 4;            // [G,8]
  float* gn_sum = ge_sum + (size_t)G * 8;       // [G,16]
  float* ecnt   = gn_sum + (size_t)G * 16;      // [G]
  float* ncnt   = ecnt + G;                     // [G]

  dim3 B(256);
  auto nb = [](long x) { return dim3((unsigned)((x + 255) / 256)); };

  // invariants + zero accumulators (workspace arrives poisoned)
  k_zero<<<nb(N), B, 0, stream>>>(deg, N);
  k_zero<<<nb((long)G * 30), B, 0, stream>>>(g, (long)G * 30);   // g..ncnt contiguous
  long EN = (E > N) ? E : N;
  k_precompute<<<nb(EN), B, 0, stream>>>(senders, receivers, node_graph,
                                         egraph, deg, ecnt, ncnt, E, N);

  // encoders
  k_edge_encoder<<<nb(E), B, 0, stream>>>(edges, We1, be1, We2, be2, e, E);
  long tiles = ((long)N + 15) / 16;
  k_node_encoder<<<dim3((unsigned)((tiles + 7) / 8)), B, 0, stream>>>(
      nodes, Wn1, bn1, Wn2, bn2, n, N);
  k_node_gsum<<<nb(N), B, 0, stream>>>(n, node_graph, gn_sum, N);
  k_g_init<<<1, 256, 0, stream>>>(gn_sum, ncnt, Wg2, bg2, g);

  // message-passing hops
  for (int hop = 0; hop < 3; hop++) {
    k_zero<<<nb((long)N * 8), B, 0, stream>>>(inc, (long)N * 8);
    k_zero<<<nb((long)G * 24), B, 0, stream>>>(ge_sum, (long)G * 24);  // ge_sum+gn_sum
    k_edge_update<<<nb(E), B, 0, stream>>>(senders, receivers, egraph, n, g,
                                           He_e, He_s, He_g, He_b,
                                           e, inc, ge_sum, E);
    k_node_update<<<nb(N), B, 0, stream>>>(node_graph, inc, deg, g,
                                           Hn_n, Hn_in, Hn_g, Hn_b,
                                           n, gn_sum, N);
    k_global_update<<<1, 256, 0, stream>>>(ge_sum, gn_sum, ecnt, ncnt,
                                           Hg_e, Hg_n, Hg_g, Hg_b, g);
  }

  // readouts -> d_out = [nodes_out (N) | globals_out (G)]
  k_readout<<<nb((long)N + G), B, 0, stream>>>(n, g, Rn, Rn_b, Rg, Rg_b,
                                               (float*)d_out, N);
  (void)n_in; (void)out_size; (void)ws_size;
}
